// LSTMCell_16346645529184
// MI455X (gfx1250) — matrix operand order repaired and checked by reading, not compiled, not run on hardware
//
#include <hip/hip_runtime.h>

// ---------------------------------------------------------------------------
// Fused LSTM cell for MI455X (gfx1250, wave32, WMMA).
//   z = x@Wx + h@Wh + bx + bh  (one K=1024 GEMM, bf16 WMMA, f32 accum)
//   i,f,o = sigmoid; g = tanh; C' = f*C + i*g; h' = o*tanh(C')
//
// Fast path (ws >= 4MB): weights pre-transposed/converted once to bf16
//   Wpk[n=2048][k=1024] in workspace (L2-resident); main loop streams x|h,
//   converts A to bf16 in LDS, copies bf16 W tiles straight to LDS, and runs
//   16 v_wmma_f32_16x16x32_bf16 per wave per K-chunk with double-buffered LDS
//   (one barrier per chunk).
// Fallback path: self-converting kernel (W fetched as f32 each chunk).
// ---------------------------------------------------------------------------

#define B_ROWS   32768
#define D_IN     512
#define D_H      512
#define NGATE_W  2048     // 4*D_H
#define K_TOT    1024     // D_IN + D_H

#define BM  128           // rows per workgroup
#define BN  64            // output cols per workgroup (per gate)
#define KC  32            // K-chunk (one bf16 WMMA of K)
#define LDA 40            // padded LDS row stride (elements): 16B-aligned rows,
                          // conflict-free ds_load_b128 fragment reads

typedef __attribute__((ext_vector_type(16))) __bf16 v16bf;
typedef __attribute__((ext_vector_type(8)))  __bf16 v8bf;
typedef __attribute__((ext_vector_type(4)))  __bf16 v4bf;
typedef __attribute__((ext_vector_type(2)))  __bf16 v2bf;
typedef __attribute__((ext_vector_type(8)))  float  v8f;

union Frag { v16bf v; v8bf h[2]; };

__device__ __forceinline__ float sigm(float z) {
  const float e = __expf(-z);
  return __fdividef(1.0f, 1.0f + e);
}
__device__ __forceinline__ float tanh_fast(float z) {
  const float e = __expf(2.0f * z);       // tanh = 1 - 2/(e^{2z}+1); exact limits
  return 1.0f - __fdividef(2.0f, e + 1.0f);
}

// ---------------------------------------------------------------------------
// One-shot weight pack: Wpk[n][k] (bf16), n in [0,2048) z-column, k in [0,1024)
// (k<512 -> Wx[k][n], else Wh[k-512][n]). 32x32 LDS transpose, coalesced R/W.
// ---------------------------------------------------------------------------
__global__ __launch_bounds__(256)
void pack_weights(const float* __restrict__ Wx, const float* __restrict__ Wh,
                  __bf16* __restrict__ Wpk)
{
  __shared__ float tile[32][33];
  const int nb = blockIdx.x * 32;          // 64 blocks over n
  const int kb = blockIdx.y * 32;          // 32 blocks over k (chunk within Wx|Wh)
  const float* Wsrc = (kb < D_IN) ? Wx : Wh;
  const int kk = kb & (D_IN - 1);
  const int tx = threadIdx.x & 31;
  const int ty = threadIdx.x >> 5;         // 0..7
#pragma unroll
  for (int r = 0; r < 32; r += 8)          // read coalesced along n
    tile[ty + r][tx] = Wsrc[(size_t)(kk + ty + r) * NGATE_W + nb + tx];
  __syncthreads();
#pragma unroll
  for (int r = 0; r < 32; r += 8)          // write coalesced along k
    Wpk[(size_t)(nb + ty + r) * K_TOT + kb + tx] = (__bf16)tile[tx][ty + r];
}

// ---------------------------------------------------------------------------
// Fast path: packed bf16 weights, double-buffered LDS, 1 barrier per chunk.
// ---------------------------------------------------------------------------
__global__ __launch_bounds__(256)
void lstm_cell_wmma_pk(const float* __restrict__ x,  const float* __restrict__ Cin,
                       const float* __restrict__ h,
                       const __bf16* __restrict__ Wpk,
                       const float* __restrict__ bx, const float* __restrict__ bh,
                       float* __restrict__ Cout, float* __restrict__ Hout)
{
  __shared__ __bf16 As[2][BM * LDA];        // 2 x 10 KB
  __shared__ __bf16 Bs[2][4 * BN * LDA];    // 2 x 20 KB

  const int tid  = threadIdx.x;
  const int lane = tid & 31;
  const int wave = tid >> 5;

  const int m_base = blockIdx.x * BM;       // 256 blocks
  const int n_base = blockIdx.y * BN;       // 8 blocks

  const int m0 = (wave & 3) * 32;           // wave row band (2 m-tiles)
  const int c0 = (wave >> 2) * 32;          // wave col band (2 n-tiles)

  v8f acc[2][4][2];
#pragma unroll
  for (int mt = 0; mt < 2; ++mt)
#pragma unroll
    for (int g = 0; g < 4; ++g)
#pragma unroll
      for (int nt = 0; nt < 2; ++nt)
#pragma unroll
        for (int v = 0; v < 8; ++v) acc[mt][g][nt][v] = 0.0f;

  // staging assignments
  const int a_row = tid >> 3;               // 0..31 (+32*r)
  const int a_kg  = tid & 7;                // k-group of 4 floats
  const int b_ng  = (tid >> 6) * D_H + n_base + (tid & 63);  // W row (z column)

  // fragment addressing (ISA 7.12.2)
  const int frow = lane & 15;
  const int ak   = (lane >> 4) * 8;         // A: lanes<16 K0-7/16-23; >=16 K8-15/24-31
  const int bk   = (lane >> 4) * 8;         // B: lanes<16 K0-7/16-23; >=16 K8-15/24-31

  float4 a_st[4];
  v8bf   b_st[4];

  auto stage = [&](int kt) {
    const int kg = kt * KC;
    const float* Ap = (kg < D_IN) ? x : h;
    const int kk = kg & (D_IN - 1);
#pragma unroll
    for (int r = 0; r < 4; ++r)
      a_st[r] = *(const float4*)(Ap + (size_t)(m_base + a_row + 32 * r) * D_IN
                                 + kk + a_kg * 4);
#pragma unroll
    for (int j = 0; j < 4; ++j)             // 64B of packed bf16 W per thread
      b_st[j] = *(const v8bf*)(Wpk + (size_t)b_ng * K_TOT + kg + j * 8);
  };

  auto commit = [&](int buf) {
#pragma unroll
    for (int r = 0; r < 4; ++r) {
      v4bf v = { (__bf16)a_st[r].x, (__bf16)a_st[r].y,
                 (__bf16)a_st[r].z, (__bf16)a_st[r].w };
      *(v4bf*)&As[buf][(a_row + 32 * r) * LDA + a_kg * 4] = v;
    }
#pragma unroll
    for (int j = 0; j < 4; ++j)
      *(v8bf*)&Bs[buf][tid * LDA + j * 8] = b_st[j];
  };

  stage(0);
  commit(0);
  int buf = 0;

  for (int kt = 0; kt < K_TOT / KC; ++kt) {
    __syncthreads();                        // buf ready for everyone
    if (kt + 1 < K_TOT / KC) stage(kt + 1); // overlap global latency w/ compute

    Frag afr[2];
#pragma unroll
    for (int mt = 0; mt < 2; ++mt) {
      const __bf16* base = &As[buf][(m0 + mt * 16 + frow) * LDA + ak];
      afr[mt].h[0] = *(const v8bf*)(base);          // ds_load_b128
      afr[mt].h[1] = *(const v8bf*)(base + 16);
    }
#pragma unroll
    for (int g = 0; g < 4; ++g) {
#pragma unroll
      for (int nt = 0; nt < 2; ++nt) {
        Frag bfr;
        const __bf16* base = &Bs[buf][(g * BN + c0 + nt * 16 + frow) * LDA + bk];
        bfr.h[0] = *(const v8bf*)(base);
        bfr.h[1] = *(const v8bf*)(base + 16);
#pragma unroll
        for (int mt = 0; mt < 2; ++mt) {
          acc[mt][g][nt] = __builtin_amdgcn_wmma_f32_16x16x32_bf16(
              false, afr[mt].v, false, bfr.v, (short)0, acc[mt][g][nt],
              false, false);
        }
      }
    }

    if (kt + 1 < K_TOT / KC) commit(buf ^ 1);  // other buffer: no hazard
    buf ^= 1;
  }

  // ---- fused gate epilogue: C/D layout lane->(N=lane%16, M=v+8*(lane/16)) ----
#pragma unroll
  for (int nt = 0; nt < 2; ++nt) {
    const int cg = n_base + c0 + nt * 16 + frow;   // col in [0, 512)
    const float bi  = bx[cg]            + bh[cg];
    const float bff = bx[D_H + cg]      + bh[D_H + cg];
    const float bo  = bx[2 * D_H + cg]  + bh[2 * D_H + cg];
    const float bg  = bx[3 * D_H + cg]  + bh[3 * D_H + cg];
#pragma unroll
    for (int mt = 0; mt < 2; ++mt) {
      const int rbase = m_base + m0 + mt * 16 + ((lane >> 4) << 3);
#pragma unroll
      for (int v = 0; v < 8; ++v) {
        const size_t idx = (size_t)(rbase + v) * D_H + cg;
        const float ig = sigm(acc[mt][0][nt][v] + bi);
        const float fg = sigm(acc[mt][1][nt][v] + bff);
        const float og = sigm(acc[mt][2][nt][v] + bo);
        const float gg = tanh_fast(acc[mt][3][nt][v] + bg);
        const float Cold = __builtin_nontemporal_load(&Cin[idx]);
        const float Cn = fg * Cold + ig * gg;
        const float hn = og * tanh_fast(Cn);
        __builtin_nontemporal_store(Cn, &Cout[idx]);   // keep L2 for weights
        __builtin_nontemporal_store(hn, &Hout[idx]);
      }
    }
  }
}

// ---------------------------------------------------------------------------
// Fallback path (ws too small): converts f32 weights per chunk.
// ---------------------------------------------------------------------------
__global__ __launch_bounds__(256)
void lstm_cell_wmma(const float* __restrict__ x,  const float* __restrict__ Cin,
                    const float* __restrict__ h,  const float* __restrict__ Wx,
                    const float* __restrict__ bx, const float* __restrict__ Wh,
                    const float* __restrict__ bh,
                    float* __restrict__ Cout, float* __restrict__ Hout)
{
  __shared__ __bf16 As[BM * LDA];
  __shared__ __bf16 Bs[4 * BN * LDA];

  const int tid  = threadIdx.x;
  const int lane = tid & 31;
  const int wave = tid >> 5;

  const int m_base = blockIdx.x * BM;
  const int n_base = blockIdx.y * BN;

  const int m0 = (wave & 3) * 32;
  const int c0 = (wave >> 2) * 32;

  v8f acc[2][4][2];
#pragma unroll
  for (int mt = 0; mt < 2; ++mt)
#pragma unroll
    for (int g = 0; g < 4; ++g)
#pragma unroll
      for (int nt = 0; nt < 2; ++nt)
#pragma unroll
        for (int v = 0; v < 8; ++v) acc[mt][g][nt][v] = 0.0f;

  const int a_row  = tid >> 3;
  const int a_kg   = tid & 7;
  const int b_ng   = (tid >> 6) * D_H + n_base + (tid & 63);

  const int frow = lane & 15;
  const int ak   = (lane >> 4) * 8;
  const int bk   = (lane >> 4) * 8;

  auto stage_load = [&](int kt, float4 a_stg[4], float b_stg[KC]) {
    const int kg = kt * KC;
    const float* Ap = (kg < D_IN) ? x  : h;
    const float* Wp = (kg < D_IN) ? Wx : Wh;
    const int kk = kg & (D_IN - 1);
#pragma unroll
    for (int r = 0; r < 4; ++r)
      a_stg[r] = *(const float4*)(Ap + (size_t)(m_base + a_row + 32 * r) * D_IN
                                  + kk + a_kg * 4);
#pragma unroll
    for (int j = 0; j < KC; ++j)
      b_stg[j] = Wp[(size_t)(kk + j) * NGATE_W + b_ng];
  };

  float4 a_st[4];
  float  b_st[KC];
  stage_load(0, a_st, b_st);

  for (int kt = 0; kt < K_TOT / KC; ++kt) {
    __syncthreads();
#pragma unroll
    for (int r = 0; r < 4; ++r) {
      v4bf v = { (__bf16)a_st[r].x, (__bf16)a_st[r].y,
                 (__bf16)a_st[r].z, (__bf16)a_st[r].w };
      *(v4bf*)&As[(a_row + 32 * r) * LDA + a_kg * 4] = v;
    }
#pragma unroll
    for (int j = 0; j < KC / 2; ++j) {
      v2bf v = { (__bf16)b_st[2 * j], (__bf16)b_st[2 * j + 1] };
      *(v2bf*)&Bs[tid * LDA + 2 * j] = v;
    }
    __syncthreads();

    if (kt + 1 < K_TOT / KC) stage_load(kt + 1, a_st, b_st);

    Frag afr[2];
#pragma unroll
    for (int mt = 0; mt < 2; ++mt) {
      const __bf16* base = &As[(m0 + mt * 16 + frow) * LDA + ak];
      afr[mt].h[0] = *(const v8bf*)(base);
      afr[mt].h[1] = *(const v8bf*)(base + 16);
    }
#pragma unroll
    for (int g = 0; g < 4; ++g) {
#pragma unroll
      for (int nt = 0; nt < 2; ++nt) {
        Frag bfr;
        const __bf16* base = &Bs[(g * BN + c0 + nt * 16 + frow) * LDA + bk];
        bfr.h[0] = *(const v8bf*)(base);
        bfr.h[1] = *(const v8bf*)(base + 16);
#pragma unroll
        for (int mt = 0; mt < 2; ++mt) {
          acc[mt][g][nt] = __builtin_amdgcn_wmma_f32_16x16x32_bf16(
              false, afr[mt].v, false, bfr.v, (short)0, acc[mt][g][nt],
              false, false);
        }
      }
    }
  }

#pragma unroll
  for (int nt = 0; nt < 2; ++nt) {
    const int cg = n_base + c0 + nt * 16 + frow;
    const float bi  = bx[cg]            + bh[cg];
    const float bff = bx[D_H + cg]      + bh[D_H + cg];
    const float bo  = bx[2 * D_H + cg]  + bh[2 * D_H + cg];
    const float bg  = bx[3 * D_H + cg]  + bh[3 * D_H + cg];
#pragma unroll
    for (int mt = 0; mt < 2; ++mt) {
      const int rbase = m_base + m0 + mt * 16 + ((lane >> 4) << 3);
#pragma unroll
      for (int v = 0; v < 8; ++v) {
        const size_t idx = (size_t)(rbase + v) * D_H + cg;
        const float ig = sigm(acc[mt][0][nt][v] + bi);
        const float fg = sigm(acc[mt][1][nt][v] + bff);
        const float og = sigm(acc[mt][2][nt][v] + bo);
        const float gg = tanh_fast(acc[mt][3][nt][v] + bg);
        const float Cold = __builtin_nontemporal_load(&Cin[idx]);
        const float Cn = fg * Cold + ig * gg;
        const float hn = og * tanh_fast(Cn);
        __builtin_nontemporal_store(Cn, &Cout[idx]);
        __builtin_nontemporal_store(hn, &Hout[idx]);
      }
    }
  }
}

extern "C" void kernel_launch(void* const* d_in, const int* in_sizes, int n_in,
                              void* d_out, int out_size, void* d_ws, size_t ws_size,
                              hipStream_t stream) {
  (void)in_sizes; (void)n_in; (void)out_size;
  const float* x  = (const float*)d_in[0];
  const float* C  = (const float*)d_in[1];
  const float* h  = (const float*)d_in[2];
  const float* Wx = (const float*)d_in[3];
  const float* bx = (const float*)d_in[4];
  const float* Wh = (const float*)d_in[5];
  const float* bh = (const float*)d_in[6];
  float* Cout = (float*)d_out;
  float* Hout = Cout + (size_t)B_ROWS * D_H;

  const dim3 grid(B_ROWS / BM, D_H / BN);   // 256 x 8 = 2048 workgroups
  const size_t pk_bytes = (size_t)NGATE_W * K_TOT * sizeof(unsigned short); // 4 MB

  if (ws_size >= pk_bytes) {
    __bf16* Wpk = (__bf16*)d_ws;
    pack_weights<<<dim3(NGATE_W / 32, K_TOT / 32), dim3(256), 0, stream>>>(Wx, Wh, Wpk);
    lstm_cell_wmma_pk<<<grid, dim3(256), 0, stream>>>(x, C, h, Wpk, bx, bh,
                                                      Cout, Hout);
  } else {
    lstm_cell_wmma<<<grid, dim3(256), 0, stream>>>(x, C, h, Wx, bx, Wh, bh,
                                                   Cout, Hout);
  }
}
